// DeformConvOriginal_20109036880555
// MI455X (gfx1250) — matrix operand order repaired and checked by reading, not compiled, not run on hardware
//
#include <hip/hip_runtime.h>
#include <hip/hip_bf16.h>

typedef _Float16 v16h __attribute__((ext_vector_type(16)));
typedef float    v8f  __attribute__((ext_vector_type(8)));
typedef uint32_t v4u  __attribute__((ext_vector_type(4)));

constexpr int Bn = 8, Cc = 64, Hh = 128, Ww = 128, Oo = 64, K2c = 9;
constexpr int HW = Hh * Ww;

// ---------------------------------------------------------------------------
// Stage 1: offset 1x1 conv: off[b][j][h][w] = sum_c x[b][c][h][w]*w_off[j][c] + b_off[j]
// ---------------------------------------------------------------------------
__global__ __launch_bounds__(256) void dc_offsets(const float* __restrict__ x,
                                                  const float* __restrict__ w_off,
                                                  const float* __restrict__ b_off,
                                                  float* __restrict__ off) {
    int pid = blockIdx.x * 256 + threadIdx.x;         // 0 .. B*HW-1
    int b   = pid >> 14;
    int rem = pid & (HW - 1);
    float acc[18];
#pragma unroll
    for (int j = 0; j < 18; ++j) acc[j] = b_off[j];
    const float* xp = x + ((size_t)b * Cc) * HW + rem;
    for (int c = 0; c < Cc; ++c) {
        float xv = xp[(size_t)c * HW];
#pragma unroll
        for (int j = 0; j < 18; ++j) acc[j] = fmaf(xv, w_off[j * Cc + c], acc[j]);
    }
#pragma unroll
    for (int j = 0; j < 18; ++j) off[((size_t)(b * 18 + j)) * HW + rem] = acc[j];
}

// ---------------------------------------------------------------------------
// Stage 2: pack weights (O,C,3,3) f32 -> f16 in per-lane fragment-contiguous
// layout [k2][cb][half][n][v2][cl]:  each lane's B fragment = 32 contiguous
// bytes (2x global_load_b128), half-wave = 512B contiguous.
//   cp = cb*16 + half*8 + v2 ; c = 2*cp + cl ; o = n
// ---------------------------------------------------------------------------
__global__ __launch_bounds__(256) void dc_packw(const float* __restrict__ w,
                                                _Float16* __restrict__ wp) {
    int d = blockIdx.x * 256 + threadIdx.x;
    if (d >= Oo * Cc * K2c) return;
    int cl  = d & 1;
    int v2  = (d >> 1) & 7;
    int n   = (d >> 4) & 63;
    int hlf = (d >> 10) & 1;
    int cb  = (d >> 11) & 1;
    int k2  = d >> 12;
    int c   = (cb * 16 + hlf * 8 + v2) * 2 + cl;
    wp[d] = (_Float16)w[(n * Cc + c) * K2c + k2];
}

// ---------------------------------------------------------------------------
// Stage 3: fused bilinear sampling + WMMA GEMM.
// Workgroup = 128 threads (4 waves), tile = 16 pixels x 64 out-channels.
// K dim = k2*64 + c (k2-major) -> each K=32 chunk has fixed k2.
// ---------------------------------------------------------------------------
__global__ __launch_bounds__(128) void dc_main(const float* __restrict__ x,
                                               const float* __restrict__ off,
                                               const _Float16* __restrict__ wp,
                                               float* __restrict__ out) {
    __shared__ __align__(32) _Float16 lds_a[2][32 * 16];  // double-buffered A tile
    __shared__ float lds_c[16][Oo];                       // output transpose tile

    const int t   = threadIdx.x;
    const int l   = t & 31;       // lane in wave
    const int wv  = t >> 5;       // wave id 0..3  -> out-channel tile
    const int m   = l & 15;       // pixel row within tile / column id
    const int hlf = l >> 4;       // lane half

    const int wg = blockIdx.x;            // B * H * (W/16) workgroups
    const int wt = wg & 7;                // wo tile
    const int ho = (wg >> 3) & 127;
    const int b  = wg >> 10;
    const int wo = wt * 16 + m;

    const uint32_t* wp32 = (const uint32_t*)wp;
    const int s0 = wv * 4;                // A-staging slots owned by this thread
    const int n  = wv * 16 + m;           // this lane's output channel

    v8f acc0 = {};                        // independent accumulators: no serial
    v8f acc1 = {};                        // WMMA->WMMA RAW chain

    for (int k2 = 0; k2 < 9; ++k2) {
        // ---- per-(pixel, k2) bilinear setup (reused for all 64 channels) ----
        const int pix = ho * Ww + wo;
        const float offy = off[((size_t)(b * 18 + 2 * k2    )) * HW + pix];
        const float offx = off[((size_t)(b * 18 + 2 * k2 + 1)) * HW + pix];
        const float py = (float)(ho - 1 + k2 / 3) + offy;
        const float px = (float)(wo - 1 + k2 % 3) + offx;
        const float y0f = floorf(py), x0f = floorf(px);
        const float ly = py - y0f, lx = px - x0f;
        const float hy = 1.0f - ly, hx = 1.0f - lx;
        const int y0 = (int)y0f, x0i = (int)x0f;
        const int y1 = y0 + 1, x1 = x0i + 1;
        const bool vy0 = (unsigned)y0  < (unsigned)Hh;
        const bool vy1 = (unsigned)y1  < (unsigned)Hh;
        const bool vx0 = (unsigned)x0i < (unsigned)Ww;
        const bool vx1 = (unsigned)x1  < (unsigned)Ww;
        const int yc0 = min(max(y0, 0), Hh - 1), yc1 = min(max(y1, 0), Hh - 1);
        const int xc0 = min(max(x0i, 0), Ww - 1), xc1 = min(max(x1, 0), Ww - 1);
        const float w00 = (vy0 && vx0) ? hy * hx : 0.0f;
        const float w01 = (vy0 && vx1) ? hy * lx : 0.0f;
        const float w10 = (vy1 && vx0) ? ly * hx : 0.0f;
        const float w11 = (vy1 && vx1) ? ly * lx : 0.0f;
        const int a00 = yc0 * Ww + xc0, a01 = yc0 * Ww + xc1;
        const int a10 = yc1 * Ww + xc0, a11 = yc1 * Ww + xc1;

#pragma unroll
        for (int cb = 0; cb < 2; ++cb) {
            const int c_base = cb * 32;
            const int buf = (k2 * 2 + cb) & 1;

            // ---- cooperatively stage 16x32 sampled A tile in fragment layout ----
            union { _Float16 h[4]; unsigned long long u; } pk;
#pragma unroll
            for (int i = 0; i < 4; ++i) {
                const int s = s0 + i;
                // A-fragment layout: lane half h, f16 slot s holds K = 16*h + s
                const int c = c_base + 16 * hlf + s;
                const float* bp = x + (((size_t)(b * Cc + c)) << 14);
                const float v = w00 * bp[a00] + w01 * bp[a01] +
                                w10 * bp[a10] + w11 * bp[a11];
                pk.h[i] = (_Float16)v;
            }
            *(unsigned long long*)&lds_a[buf][l * 16 + s0] = pk.u;
            __syncthreads();  // double-buffering: one barrier per chunk suffices

            // ---- A fragment (shared by all 4 waves) ----
            v16h afrag = *(const v16h*)&lds_a[buf][l * 16];

            // ---- B fragment: 32 contiguous bytes per lane (2x b128) ----
            union { v16h h; v4u q[2]; } bu;
            const uint32_t* bptr =
                wp32 + (size_t)((((k2 * 2 + cb) * 2 + hlf) * 64 + n) << 3);
            bu.q[0] = *(const v4u*)(bptr);
            bu.q[1] = *(const v4u*)(bptr + 4);

            if (cb == 0)
                acc0 = __builtin_amdgcn_wmma_f32_16x16x32_f16(false, afrag, false, bu.h,
                                                              (short)0, acc0, false, false);
            else
                acc1 = __builtin_amdgcn_wmma_f32_16x16x32_f16(false, afrag, false, bu.h,
                                                              (short)0, acc1, false, false);
        }
    }

    const v8f acc = acc0 + acc1;

    // ---- transpose through LDS for coalesced global stores ----
    {
#pragma unroll
        for (int r = 0; r < 8; ++r) lds_c[r + 8 * hlf][n] = acc[r];
    }
    __syncthreads();
    {
        const int no = t >> 1;
        const int m8 = (t & 1) * 8;
        float* op = out + (((size_t)(b * Oo + no)) * Hh + ho) * Ww + wt * 16 + m8;
#pragma unroll
        for (int r = 0; r < 8; ++r) op[r] = lds_c[m8 + r][no];
    }
}

// ---------------------------------------------------------------------------
extern "C" void kernel_launch(void* const* d_in, const int* in_sizes, int n_in,
                              void* d_out, int out_size, void* d_ws, size_t ws_size,
                              hipStream_t stream) {
    const float* x      = (const float*)d_in[0];
    const float* weight = (const float*)d_in[1];
    const float* w_off  = (const float*)d_in[2];
    const float* b_off  = (const float*)d_in[3];
    float* out = (float*)d_out;

    float*    off   = (float*)d_ws;                                          // 9.44 MB
    _Float16* wpack = (_Float16*)((char*)d_ws + (size_t)Bn * 18 * HW * sizeof(float));

    dc_offsets<<<(Bn * HW) / 256, 256, 0, stream>>>(x, w_off, b_off, off);
    dc_packw<<<(Oo * Cc * K2c + 255) / 256, 256, 0, stream>>>(weight, wpack);
    dc_main<<<Bn * Hh * (Ww / 16), 128, 0, stream>>>(x, off, wpack, out);
}
